// TrajNet_77936476553902
// MI455X (gfx1250) — matrix operand order repaired and checked by reading, not compiled, not run on hardware
//
#include <hip/hip_runtime.h>
#include <hip/hip_bf16.h>
#include <math.h>

typedef __bf16 bf16_t;
typedef __bf16 v16bf __attribute__((ext_vector_type(16)));
typedef float  v8f   __attribute__((ext_vector_type(8)));

#define S_DIM   1024
#define H_DIM   4096
#define NA      4
#define MAXT    511
#define TP1     512
#define BATCH   64
#define ROWS    (BATCH * TP1)          // 32768
#define MTILE   16
#define NTILES  (H_DIM / 16)           // 256
#define KTILES  (S_DIM / 32)           // 32
#define FRAG_ELEMS 512                 // 32x16 bf16 per fragment
#define W1PACK_ELEMS ((size_t)NTILES * KTILES * FRAG_ELEMS)  // 4,194,304

// LDS layout (bytes)
#define SA_PITCH     1032                      // bf16 elems per row (pad: conflict-free b128)
#define SA_BYTES     (MTILE * SA_PITCH * 2)    // 33024
#define H_PITCH      17                        // floats per n (pad)
#define H_BYTES      (H_DIM * H_PITCH * 4)     // 278528
#define ZPART_OFF    (SA_BYTES + H_BYTES)      // 311552
#define ZFULL_OFF    (ZPART_OFF + 256 * 4)
#define CARR_OFF     (ZFULL_OFF + 64 * 4)
#define SMEM_BYTES   (CARR_OFF + 16 * 4)       // 312896 < 320KB

// ---------------------------------------------------------------------------
// Kernel 0: repack W1 (S x H, fp32 row-major) into bf16 WMMA B-fragments.
// Fragment (nTile,kTile): 512 bf16, elem index = lane*16 + e,
//   n = nTile*16 + (lane&15),  K = kTile*32 + (e>>3)*16 + (lane>>4)*8 + (e&7).
// ---------------------------------------------------------------------------
__global__ void pack_w1(const float* __restrict__ W1, bf16_t* __restrict__ out) {
    size_t f = (size_t)blockIdx.x * blockDim.x + threadIdx.x;
    if (f >= W1PACK_ELEMS) return;
    int e     = (int)(f & 15);
    int lane  = (int)((f >> 4) & 31);
    int kTile = (int)((f >> 9) & (KTILES - 1));
    int nTile = (int)(f >> 14);
    int K = kTile * 32 + ((e >> 3) << 4) + ((lane >> 4) << 3) + (e & 7);
    int n = (nTile << 4) + (lane & 15);
    out[f] = (bf16_t)W1[(size_t)K * H_DIM + n];
}

// ---------------------------------------------------------------------------
// Kernel 1: fused GEMM1(bf16 WMMA) + bias/ReLU + 4-col GEMM2 + log-softmax
//           + gather + mask, one 16-row tile per workgroup.
// Phase 1 processes two N-tiles per K-sweep: shared A fragment (halves LDS
// traffic) and two independent WMMA accumulator chains (hides XDL latency).
// ---------------------------------------------------------------------------
__global__ void __launch_bounds__(256)
traj_main(const float* __restrict__ s, const int* __restrict__ actions,
          const int* __restrict__ lengths, const bf16_t* __restrict__ w1p,
          const float* __restrict__ b1, const float* __restrict__ W2,
          const float* __restrict__ b2, float* __restrict__ partials)
{
    extern __shared__ __align__(16) char smem[];
    bf16_t* sA    = (bf16_t*)smem;
    float*  hbuf  = (float*)(smem + SA_BYTES);
    float*  zpart = (float*)(smem + ZPART_OFF);
    float*  zfull = (float*)(smem + ZFULL_OFF);
    float*  carr  = (float*)(smem + CARR_OFF);

    const int tid  = threadIdx.x;
    const int wave = tid >> 5;
    const int lane = tid & 31;
    const int g    = lane >> 4;   // half-group
    const int nl   = lane & 15;
    const size_t rowBase = (size_t)blockIdx.x * MTILE;

    // ---- Phase 0: stage 16 contiguous rows of s as bf16 into LDS ----------
    {
        const float* src = s + rowBase * S_DIM;
        for (int i = tid; i < MTILE * S_DIM; i += 256) {
            int r = i >> 10, c = i & (S_DIM - 1);
            sA[r * SA_PITCH + c] = (bf16_t)src[i];
        }
    }
    __syncthreads();

    // ---- Phase 1: h = relu(s @ W1 + b1), each wave owns 512 H-columns ----
    const char* abase = (const char*)sA + (size_t)nl * (SA_PITCH * 2) + g * 16;
    for (int jt = 0; jt < 32; jt += 2) {
        const int nTile0 = wave * 32 + jt;
        v8f acc0 = {}, acc1 = {};
        const bf16_t* bsrc0 = w1p + (size_t)nTile0 * KTILES * FRAG_ELEMS + lane * 16;
        const bf16_t* bsrc1 = bsrc0 + (size_t)KTILES * FRAG_ELEMS;
        #pragma unroll 4
        for (int kt = 0; kt < KTILES; ++kt) {
            union { v16bf v; uint4 u[2]; } A, B0, B1;
            const uint4* ap = (const uint4*)(abase + kt * 64);
            A.u[0] = ap[0];            // K = kt*32 + g*8 + [0..7]
            A.u[1] = ap[2];            // K = kt*32 + g*8 + 16 + [0..7]
            const uint4* bp0 = (const uint4*)(bsrc0 + (size_t)kt * FRAG_ELEMS);
            const uint4* bp1 = (const uint4*)(bsrc1 + (size_t)kt * FRAG_ELEMS);
            B0.u[0] = bp0[0];
            B0.u[1] = bp0[1];
            B1.u[0] = bp1[0];
            B1.u[1] = bp1[1];
            acc0 = __builtin_amdgcn_wmma_f32_16x16x32_bf16(
                false, A.v, false, B0.v, (short)0, acc0, false, false);
            acc1 = __builtin_amdgcn_wmma_f32_16x16x32_bf16(
                false, A.v, false, B1.v, (short)0, acc1, false, false);
        }
        // Epilogue for both N-tiles: +b1, ReLU, store to LDS h (col-major).
        #pragma unroll
        for (int p = 0; p < 2; ++p) {
            const v8f acc = p ? acc1 : acc0;
            const int ncol = (nTile0 + p) * 16 + nl;
            const float bias = b1[ncol];
            float4 lo, hi;
            lo.x = fmaxf(acc[0] + bias, 0.f); lo.y = fmaxf(acc[1] + bias, 0.f);
            lo.z = fmaxf(acc[2] + bias, 0.f); lo.w = fmaxf(acc[3] + bias, 0.f);
            hi.x = fmaxf(acc[4] + bias, 0.f); hi.y = fmaxf(acc[5] + bias, 0.f);
            hi.z = fmaxf(acc[6] + bias, 0.f); hi.w = fmaxf(acc[7] + bias, 0.f);
            // lane holds rows g*8 + [0..7] of column ncol (C-layout)
            float* hp = hbuf + (size_t)ncol * H_PITCH + g * 8;
            *(float4*)(hp + 0) = lo;
            *(float4*)(hp + 4) = hi;
        }
    }
    __syncthreads();

    // ---- Phase 2: z[row][a] = h[row,:] . W2[:,a] + b2[a] ------------------
    {
        const int combo = tid & 63;          // (row, a)
        const int chunk = tid >> 6;          // H quarter
        const int row = combo >> 2, a = combo & 3;
        float acc2 = 0.f;
        const int n0 = chunk * (H_DIM / 4), n1 = n0 + (H_DIM / 4);
        #pragma unroll 4
        for (int n = n0; n < n1; ++n)
            acc2 += hbuf[(size_t)n * H_PITCH + row] * W2[(size_t)n * 32 + a];
        zpart[tid] = acc2;
    }
    __syncthreads();
    if (tid < 64) {
        zfull[tid] = zpart[tid] + zpart[tid + 64] + zpart[tid + 128] +
                     zpart[tid + 192] + b2[tid & 3];
    }
    __syncthreads();

    // ---- Phase 3: log-softmax over NA=4, gather action, mask --------------
    if (tid < 16) {
        const float z0 = zfull[tid * 4 + 0], z1 = zfull[tid * 4 + 1];
        const float z2 = zfull[tid * 4 + 2], z3 = zfull[tid * 4 + 3];
        const float m  = fmaxf(fmaxf(z0, z1), fmaxf(z2, z3));
        const float lse = m + logf(expf(z0 - m) + expf(z1 - m) +
                                   expf(z2 - m) + expf(z3 - m));
        const size_t r = rowBase + tid;
        const int b = (int)(r >> 9);
        const int t = (int)(r & (TP1 - 1));
        float c = 0.f;
        if (t < MAXT && t < lengths[b]) {
            const int act = actions[b * MAXT + t];
            const float za = (act == 0) ? z0 : (act == 1) ? z1 : (act == 2) ? z2 : z3;
            c = za - lse;
        }
        carr[tid] = c;
    }
    __syncthreads();
    if (tid == 0) {
        float ssum = 0.f;
        for (int i = 0; i < 16; ++i) ssum += carr[i];
        partials[blockIdx.x] = ssum;
    }
}

// ---------------------------------------------------------------------------
// Kernel 2: deterministic fixed-order reduction of 2048 partials.
// ---------------------------------------------------------------------------
__global__ void reduce_partials(const float* __restrict__ p, float* __restrict__ out) {
    __shared__ float r[256];
    float local = 0.f;
    for (int i = threadIdx.x; i < ROWS / MTILE; i += 256) local += p[i];
    r[threadIdx.x] = local;
    __syncthreads();
    for (int sft = 128; sft > 0; sft >>= 1) {
        if ((int)threadIdx.x < sft) r[threadIdx.x] += r[threadIdx.x + sft];
        __syncthreads();
    }
    if (threadIdx.x == 0) out[0] = -r[0];
}

extern "C" void kernel_launch(void* const* d_in, const int* in_sizes, int n_in,
                              void* d_out, int out_size, void* d_ws, size_t ws_size,
                              hipStream_t stream) {
    (void)in_sizes; (void)n_in; (void)out_size; (void)ws_size;
    const float* s_i     = (const float*)d_in[0];
    const int*   actions = (const int*)d_in[1];
    const int*   lengths = (const int*)d_in[2];
    const float* W1      = (const float*)d_in[3];
    const float* b1      = (const float*)d_in[4];
    const float* W2      = (const float*)d_in[5];
    const float* b2      = (const float*)d_in[6];

    bf16_t* w1p      = (bf16_t*)d_ws;                                   // 8 MB
    float*  partials = (float*)((char*)d_ws + W1PACK_ELEMS * sizeof(bf16_t));

    pack_w1<<<(int)(W1PACK_ELEMS / 256), 256, 0, stream>>>(W1, w1p);
    traj_main<<<ROWS / MTILE, 256, SMEM_BYTES, stream>>>(
        s_i, actions, lengths, w1p, b1, W2, b2, partials);
    reduce_partials<<<1, 256, 0, stream>>>(partials, (float*)d_out);
}
